// ClusterNet_17549236371952
// MI455X (gfx1250) — matrix operand order repaired and checked by reading, not compiled, not run on hardware
//
#include <hip/hip_runtime.h>
#include <hip/hip_bf16.h>

typedef __attribute__((ext_vector_type(16))) _Float16 v16h;
typedef __attribute__((ext_vector_type(8)))  _Float16 v8h;
typedef __attribute__((ext_vector_type(4)))  _Float16 v4h;
typedef __attribute__((ext_vector_type(8)))  float    v8f;

#define BS 8192
#define NH 2048
#define NC 512

// ---------------------------------------------------------------------------
// Kernel 1: per-row squared norm of z.  8 waves/WG, one wave per row.
// ---------------------------------------------------------------------------
__global__ void k_zsq(const float* __restrict__ z, float* __restrict__ zsq) {
    const int wid  = threadIdx.x >> 5;
    const int lane = threadIdx.x & 31;
    const int row  = blockIdx.x * 8 + wid;
    const float* p = z + (size_t)row * NH;
    float s = 0.f;
#pragma unroll 8
    for (int i = 0; i < NH / 32; ++i) {
        float x = p[lane + i * 32];
        s += x * x;
    }
#pragma unroll
    for (int off = 16; off > 0; off >>= 1) s += __shfl_xor(s, off, 32);
    if (lane == 0) zsq[row] = s;
}

// ---------------------------------------------------------------------------
// Kernel 2: per-row squared norm of centroids + zero the Q-column-sum accum.
// ---------------------------------------------------------------------------
__global__ void k_csq(const float* __restrict__ c, float* __restrict__ csq,
                      float* __restrict__ cs) {
    const int wid  = threadIdx.x >> 5;
    const int lane = threadIdx.x & 31;
    const int row  = blockIdx.x * 8 + wid;
    const float* p = c + (size_t)row * NH;
    float s = 0.f;
#pragma unroll 8
    for (int i = 0; i < NH / 32; ++i) {
        float x = p[lane + i * 32];
        s += x * x;
    }
#pragma unroll
    for (int off = 16; off > 0; off >>= 1) s += __shfl_xor(s, off, 32);
    if (lane == 0) { csq[row] = s; cs[row] = 0.f; }  // re-zero every launch
}

// ---------------------------------------------------------------------------
// Kernel 3: WMMA GEMM (cross = z @ centroids^T) fused with distance ->
// Student-t transform.  Writes UNNORMALIZED Q.
//   WG: 256 threads = 8 wave32, waves tiled 4(M) x 2(N).
//   Wave tile: 64x32 -> 4x2 v_wmma_f32_16x16x32_f16 accumulators.
//   WG tile: 256(M) x 64(N).  K stepped by 32, 64 iterations.
//   Both A (z) and B (centroids) tiles staged f32->f16 through LDS,
//   double-buffered, with register prefetch of the next K-slice.
//   Rows padded to 40 halves (80 B) so 16-lane fragment reads start on
//   bank offsets spaced 4 apart -> conflict-free ds_load_b128.
// ---------------------------------------------------------------------------
__global__ void __launch_bounds__(256, 2)
k_gemm(const float* __restrict__ z, const float* __restrict__ cent,
       const float* __restrict__ zsq, const float* __restrict__ csq,
       float* __restrict__ qraw) {
    __shared__ __align__(16) _Float16 As[2][256 * 40];  // 2 x 20 KB
    __shared__ __align__(16) _Float16 Bs[2][ 64 * 40];  // 2 x  5 KB

    const int tid   = threadIdx.x;
    const int wid   = tid >> 5;
    const int lane  = tid & 31;
    const int waveM = wid & 3;   // 0..3
    const int waveN = wid >> 2;  // 0..1
    const int lhalf = lane >> 4; // 0/1 : K-half selector (ISA 7.12.2)
    const int l16   = lane & 15;

    const int M0 = blockIdx.y * 256;
    const int N0 = blockIdx.x * 64;

    float4 pa[8];   // A prefetch: 256x32 f32 / 256 thr = 8 float4 each
    float4 pb[2];   // B prefetch:  64x32 f32 / 256 thr = 2 float4 each

    auto load_tile = [&](int kb) {
#pragma unroll
        for (int j = 0; j < 8; ++j) {
            int id = tid + j * 256;              // 0..2047 float4 slots
            int r  = id >> 3;                    // A row 0..255
            int c4 = id & 7;                     // float4 col 0..7
            pa[j] = *(const float4*)(z + (size_t)(M0 + r) * NH + kb + c4 * 4);
        }
#pragma unroll
        for (int j = 0; j < 2; ++j) {
            int id = tid + j * 256;              // 0..511 float4 slots
            int r  = id >> 3;                    // B row 0..63
            int c4 = id & 7;
            pb[j] = *(const float4*)(cent + (size_t)(N0 + r) * NH + kb + c4 * 4);
        }
    };
    auto store_tile = [&](int buf) {
#pragma unroll
        for (int j = 0; j < 8; ++j) {
            int id = tid + j * 256;
            int r  = id >> 3;
            int c4 = id & 7;
            v4h h;
            h[0] = (_Float16)pa[j].x; h[1] = (_Float16)pa[j].y;
            h[2] = (_Float16)pa[j].z; h[3] = (_Float16)pa[j].w;
            *(v4h*)&As[buf][r * 40 + c4 * 4] = h;
        }
#pragma unroll
        for (int j = 0; j < 2; ++j) {
            int id = tid + j * 256;
            int r  = id >> 3;
            int c4 = id & 7;
            v4h h;
            h[0] = (_Float16)pb[j].x; h[1] = (_Float16)pb[j].y;
            h[2] = (_Float16)pb[j].z; h[3] = (_Float16)pb[j].w;
            *(v4h*)&Bs[buf][r * 40 + c4 * 4] = h;
        }
    };

    v8f acc[4][2] = {};

    load_tile(0);
    store_tile(0);
    __syncthreads();

    for (int i = 0; i < 64; ++i) {
        const int cur = i & 1;
        if (i < 63) load_tile((i + 1) * 32);     // prefetch next K-slice

        // A fragments: lane l16 = M row; lanes 0-15 hold K 0..15,
        // lanes 16-31 hold K 16..31
        v16h a[4];
#pragma unroll
        for (int s = 0; s < 4; ++s) {
            int m = waveM * 64 + s * 16 + l16;
            v8h c0 = *(const v8h*)&As[cur][m * 40 + lhalf * 16];
            v8h c1 = *(const v8h*)&As[cur][m * 40 + lhalf * 16 + 8];
            a[s] = __builtin_shufflevector(c0, c1,
                    0,1,2,3,4,5,6,7,8,9,10,11,12,13,14,15);
        }
        // B fragments: lane l16 = N col; lanes 0-15 K 0..15, 16-31 K 16..31
        v16h b[2];
#pragma unroll
        for (int t = 0; t < 2; ++t) {
            int n = waveN * 32 + t * 16 + l16;
            b[t] = *(const v16h*)&Bs[cur][n * 40 + lhalf * 16];
        }

#pragma unroll
        for (int s = 0; s < 4; ++s)
#pragma unroll
            for (int t = 0; t < 2; ++t)
                acc[s][t] = __builtin_amdgcn_wmma_f32_16x16x32_f16(
                    false, a[s], false, b[t], (short)0, acc[s][t], false, false);

        if (i < 63) store_tile(cur ^ 1);         // fill the other buffer
        __syncthreads();
    }

    // Epilogue: d2 = |z|^2 + |c|^2 - 2*cross ; Q_raw = 1/(1+sqrt(d2))
    // C/D layout: element v of v8f -> M = v + 8*lhalf, N = l16.
#pragma unroll
    for (int s = 0; s < 4; ++s) {
#pragma unroll
        for (int t = 0; t < 2; ++t) {
            int col  = N0 + waveN * 32 + t * 16 + l16;
            float cq = csq[col];
#pragma unroll
            for (int v = 0; v < 8; ++v) {
                int row = M0 + waveM * 64 + s * 16 + lhalf * 8 + v;
                float d2 = zsq[row] + cq - 2.0f * acc[s][t][v];
                d2 = d2 > 0.f ? d2 : 0.f;
                float sim = __builtin_amdgcn_sqrtf(d2);
                // ALPHA=1: (1+sim/1)^(-(1+1)/2) = 1/(1+sim)
                qraw[(size_t)row * NC + col] = __builtin_amdgcn_rcpf(1.0f + sim);
            }
        }
    }
}

// ---------------------------------------------------------------------------
// Kernel 4: row-normalize Q in place; accumulate column sums of normalized Q.
// 128 WGs x 256 threads; each wave handles 8 rows; per-WG LDS column partials
// then one global atomicAdd per column per WG.
// ---------------------------------------------------------------------------
__global__ void k_qnorm(float* __restrict__ q, float* __restrict__ cs) {
    __shared__ float csp[NC];
    const int tid  = threadIdx.x;
    const int wid  = tid >> 5;
    const int lane = tid & 31;
    csp[tid] = 0.f; csp[tid + 256] = 0.f;
    __syncthreads();

    for (int r = 0; r < 8; ++r) {
        int row = blockIdx.x * 64 + wid * 8 + r;
        float* qp = q + (size_t)row * NC;
        float vals[16];
        float s = 0.f;
#pragma unroll
        for (int i = 0; i < 16; ++i) {
            vals[i] = qp[lane + i * 32];
            s += vals[i];
        }
#pragma unroll
        for (int off = 16; off > 0; off >>= 1) s += __shfl_xor(s, off, 32);
        float inv = __builtin_amdgcn_rcpf(s);
#pragma unroll
        for (int i = 0; i < 16; ++i) {
            float qv = vals[i] * inv;
            qp[lane + i * 32] = qv;
            atomicAdd(&csp[lane + i * 32], qv);
        }
    }
    __syncthreads();
    atomicAdd(&cs[tid],       csp[tid]);
    atomicAdd(&cs[tid + 256], csp[tid + 256]);
}

// ---------------------------------------------------------------------------
// Kernel 5: P = rownorm( Q^2 / colsum(Q) ).  One wave per row.
// ---------------------------------------------------------------------------
__global__ void k_p(const float* __restrict__ q, const float* __restrict__ cs,
                    float* __restrict__ p) {
    const int wid  = threadIdx.x >> 5;
    const int lane = threadIdx.x & 31;
    const int row  = blockIdx.x * 8 + wid;
    const float* qp = q + (size_t)row * NC;
    float* pp = p + (size_t)row * NC;
    float vals[16];
    float s = 0.f;
#pragma unroll
    for (int i = 0; i < 16; ++i) {
        int c = lane + i * 32;
        float qv = qp[c];
        float t  = qv * qv * __builtin_amdgcn_rcpf(cs[c]);
        vals[i] = t;
        s += t;
    }
#pragma unroll
    for (int off = 16; off > 0; off >>= 1) s += __shfl_xor(s, off, 32);
    float inv = __builtin_amdgcn_rcpf(s);
#pragma unroll
    for (int i = 0; i < 16; ++i) pp[lane + i * 32] = vals[i] * inv;
}

// ---------------------------------------------------------------------------
extern "C" void kernel_launch(void* const* d_in, const int* in_sizes, int n_in,
                              void* d_out, int out_size, void* d_ws, size_t ws_size,
                              hipStream_t stream) {
    const float* z    = (const float*)d_in[0];   // (8192, 2048) f32
    const float* cent = (const float*)d_in[1];   // (512, 2048)  f32
    float* Q = (float*)d_out;                    // (8192, 512)
    float* P = Q + (size_t)BS * NC;              // (8192, 512)

    float* zsq = (float*)d_ws;      // 8192 f32
    float* csq = zsq + BS;          // 512 f32
    float* cs  = csq + NC;          // 512 f32   (~37 KB total workspace)

    k_zsq <<<BS / 8, 256, 0, stream>>>(z, zsq);
    k_csq <<<NC / 8, 256, 0, stream>>>(cent, csq, cs);

    dim3 grid(NC / 64, BS / 256);   // (8, 32)
    k_gemm<<<grid, 256, 0, stream>>>(z, cent, zsq, csq, Q);

    k_qnorm<<<BS / 64, 256, 0, stream>>>(Q, cs);
    k_p   <<<BS / 8,  256, 0, stream>>>(Q, cs, P);
}
